// Attention_47098611368058
// MI455X (gfx1250) — matrix operand order repaired and checked by reading, not compiled, not run on hardware
//
#include <hip/hip_runtime.h>
#include <hip/hip_bf16.h>

typedef __attribute__((ext_vector_type(16))) __bf16 bf16x16;
typedef __attribute__((ext_vector_type(8)))  __bf16 bf16x8;
typedef __attribute__((ext_vector_type(8)))  float  floatx8;
typedef int v4i_vs __attribute__((vector_size(16)));   // 16-byte payload type

union FragA { bf16x16 f; bf16x8 h[2]; };

__device__ __forceinline__ __bf16 f2bf(float x) { return (__bf16)x; }

// ---- CDNA5 async global->LDS copy (ASYNCcnt path), with safe fallback ------
#define HAS_ASYNC __has_builtin(__builtin_amdgcn_global_load_async_to_lds_b128)

__device__ __forceinline__ void async_cp16(const __bf16* g, __bf16* l) {
#if HAS_ASYNC
  __builtin_amdgcn_global_load_async_to_lds_b128(
      (__attribute__((address_space(1))) v4i_vs*)g,
      (__attribute__((address_space(3))) v4i_vs*)l, 0, 0);
#else
  *(bf16x8*)l = *(const bf16x8*)g;          // fallback: VGPR round-trip
#endif
}
__device__ __forceinline__ void async_wait() {
#if HAS_ASYNC
#if __has_builtin(__builtin_amdgcn_s_wait_asynccnt)
  __builtin_amdgcn_s_wait_asynccnt(0);
#else
  asm volatile("s_wait_asynccnt 0" ::: "memory");
#endif
#endif
}

// ---------------------------------------------------------------------------
// Kernel 0: one-shot precision/layout prep so every GEMM tile is a straight
// 2D bf16 copy (async-LDS friendly):
//   Xb   [4096,1024]  = bf16(x)
//   Wqt  [3072,1024]  = bf16(w_qkv^T)
//   Wot  [1024,1024]  = bf16(w_out^T)
// ---------------------------------------------------------------------------
__global__ __launch_bounds__(256) void prep(const float* __restrict__ X,
                                            const float* __restrict__ Wq,
                                            const float* __restrict__ Wo,
                                            __bf16* __restrict__ Xb,
                                            __bf16* __restrict__ Wqt,
                                            __bf16* __restrict__ Wot) {
  int i = blockIdx.x * 256 + threadIdx.x;
  if (i < 4096 * 1024) Xb[i] = f2bf(X[i]);
  if (i < 1024 * 3072) {                     // Wq[k][n] -> Wqt[n][k]
    int k = i / 3072, n = i % 3072;
    Wqt[n * 1024 + k] = f2bf(Wq[i]);
  }
  if (i < 1024 * 1024) {                     // Wo[k][n] -> Wot[n][k]
    int k = i >> 10, n = i & 1023;
    Wot[n * 1024 + k] = f2bf(Wo[i]);
  }
}

// ---------------------------------------------------------------------------
// Kernel 1: qkv = x @ w_qkv  (M=4096,K=1024,N=3072). 128x64 block tile,
// 4 waves x (32x64) strips, K-step 32, double-buffered async LDS staging.
// Scatter Q,K -> [B,H,S,Dh]; V -> [B,H,Dh,S] (pre-transposed for attention).
// ---------------------------------------------------------------------------
__global__ __launch_bounds__(128) void qkv_gemm(const __bf16* __restrict__ Xb,
                                                const __bf16* __restrict__ Wt,
                                                __bf16* __restrict__ Qo,
                                                __bf16* __restrict__ Ko,
                                                __bf16* __restrict__ Vo) {
  constexpr int Kd = 1024;
  __shared__ __bf16 lA[2][128 * 40];        // A tile row-major, stride 40
  __shared__ __bf16 lB[2][64 * 40];         // Wt rows: lB[n*40 + k]
  const int tid  = threadIdx.x;
  const int wave = tid >> 5, lane = tid & 31;
  const int g = lane >> 4, ln = lane & 15;
  const int m0 = blockIdx.x * 128, n0 = blockIdx.y * 64;

  floatx8 acc[2][4] = {};

  auto load_tiles = [&](int k0, int buf) {
    #pragma unroll
    for (int c = tid; c < 512; c += 128) {  // A: 128 rows x 4 chunks(8h)
      int r = c >> 2, cc = (c & 3) * 8;
      async_cp16(Xb + (m0 + r) * Kd + k0 + cc, &lA[buf][r * 40 + cc]);
    }
    #pragma unroll
    for (int c = tid; c < 256; c += 128) {  // B: 64 n-rows x 4 chunks
      int r = c >> 2, cc = (c & 3) * 8;
      async_cp16(Wt + (size_t)(n0 + r) * Kd + k0 + cc, &lB[buf][r * 40 + cc]);
    }
  };

  load_tiles(0, 0);
  async_wait();
  __syncthreads();

  for (int kt = 0; kt < 32; ++kt) {
    const int cur = kt & 1;
    if (kt < 31) load_tiles((kt + 1) * 32, cur ^ 1);   // async prefetch

    FragA a[2];
    #pragma unroll
    for (int u = 0; u < 2; ++u) {
      const __bf16* ar = &lA[cur][(wave * 32 + u * 16 + ln) * 40];
      a[u].h[0] = *(const bf16x8*)(ar + 8 * g);
      a[u].h[1] = *(const bf16x8*)(ar + 16 + 8 * g);
    }
    #pragma unroll
    for (int t = 0; t < 4; ++t) {
      FragA b;
      const __bf16* br = &lB[cur][(t * 16 + ln) * 40];
      b.h[0] = *(const bf16x8*)(br + 8 * g);
      b.h[1] = *(const bf16x8*)(br + 16 + 8 * g);
      #pragma unroll
      for (int u = 0; u < 2; ++u)
        acc[u][t] = __builtin_amdgcn_wmma_f32_16x16x32_bf16(
            false, a[u].f, false, b.f, (short)0, acc[u][t], false, false);
    }
    async_wait();
    __syncthreads();
  }

  #pragma unroll
  for (int u = 0; u < 2; ++u)
    #pragma unroll
    for (int t = 0; t < 4; ++t)
      #pragma unroll
      for (int v = 0; v < 8; ++v) {
        int m = m0 + wave * 32 + u * 16 + v + 8 * g;
        int e = n0 + 16 * t + ln;
        int bb = m >> 11, s = m & 2047;
        int which = e >> 10, he = e & 1023;
        int h = he >> 6, d = he & 63;
        __bf16 val = f2bf(acc[u][t][v]);
        int hh = bb * 16 + h;
        if (which == 0)      Qo[((hh * 2048) + s) * 64 + d] = val;
        else if (which == 1) Ko[((hh * 2048) + s) * 64 + d] = val;
        else                 Vo[((hh * 64) + d) * 2048 + s] = val;  // V^T
      }
}

// ---------------------------------------------------------------------------
// Kernel 2: causal flash attention. Double-buffered async K/V tiles.
// K: [B,H,S,Dh] row-major; V: [B,H,Dh,S] (already transposed).
// ---------------------------------------------------------------------------
__global__ __launch_bounds__(128) void attn_fa(const __bf16* __restrict__ Qg,
                                               const __bf16* __restrict__ Kg,
                                               const __bf16* __restrict__ Vt,
                                               __bf16* __restrict__ Og) {
  __shared__ __bf16 lK[2][64 * 72];         // lK[j*72 + d]
  __shared__ __bf16 lV[2][64 * 72];         // lV[d*72 + j]
  __shared__ __bf16 lP[4][16 * 72];         // per-wave P staging
  const int tid  = threadIdx.x;
  const int wave = tid >> 5, lane = tid & 31;
  const int g = lane >> 4, ln = lane & 15;
  const int ib = blockIdx.x;                // query row block (64 rows)
  const int bh = blockIdx.y;                // b*16 + h
  const __bf16* Qh = Qg + (size_t)bh * (2048 * 64);
  const __bf16* Kh = Kg + (size_t)bh * (2048 * 64);
  const __bf16* Vh = Vt + (size_t)bh * (64 * 2048);
  const int row0 = ib * 64 + wave * 16;

  FragA qa[2];
  {
    const __bf16* qr = Qh + (row0 + ln) * 64;
    #pragma unroll
    for (int c = 0; c < 2; ++c) {
      qa[c].h[0] = *(const bf16x8*)(qr + 32 * c + 8 * g);
      qa[c].h[1] = *(const bf16x8*)(qr + 32 * c + 16 + 8 * g);
    }
  }

  auto load_tiles = [&](int jb, int buf) {
    #pragma unroll
    for (int c = tid; c < 512; c += 128) {  // 64 rows x 8 chunks each tensor
      int r = c >> 3, cc = (c & 7) * 8;
      async_cp16(Kh + (jb * 64 + r) * 64 + cc, &lK[buf][r * 72 + cc]);
      async_cp16(Vh + r * 2048 + jb * 64 + cc, &lV[buf][r * 72 + cc]);
    }
  };

  floatx8 oacc[4] = {};
  float mrow[8], lrow[8];
  #pragma unroll
  for (int v = 0; v < 8; ++v) { mrow[v] = -1e30f; lrow[v] = 0.f; }

  load_tiles(0, 0);
  async_wait();
  __syncthreads();

  for (int jb = 0; jb <= ib; ++jb) {
    const int cur = jb & 1;
    if (jb < ib) load_tiles(jb + 1, cur ^ 1);          // async prefetch

    floatx8 sacc[4] = {};                              // S = Q K^T
    #pragma unroll
    for (int c = 0; c < 2; ++c)
      #pragma unroll
      for (int t = 0; t < 4; ++t) {
        FragA b;
        const __bf16* kr = &lK[cur][(t * 16 + ln) * 72 + 32 * c + 8 * g];
        b.h[0] = *(const bf16x8*)kr;
        b.h[1] = *(const bf16x8*)(kr + 16);
        sacc[t] = __builtin_amdgcn_wmma_f32_16x16x32_bf16(
            false, qa[c].f, false, b.f, (short)0, sacc[t], false, false);
      }

    const bool diag = (jb == ib);
    #pragma unroll
    for (int t = 0; t < 4; ++t)
      #pragma unroll
      for (int v = 0; v < 8; ++v) {
        float s = sacc[t][v] * 0.125f;
        if (diag) {
          int col = jb * 64 + 16 * t + ln;
          int row = row0 + v + 8 * g;
          if (col > row) s = -1e30f;
        }
        sacc[t][v] = s;
      }

    float mnew[8], corr[8], psum[8];
    #pragma unroll
    for (int v = 0; v < 8; ++v) {
      float r = fmaxf(fmaxf(sacc[0][v], sacc[1][v]), fmaxf(sacc[2][v], sacc[3][v]));
      #pragma unroll
      for (int off = 1; off < 16; off <<= 1)
        r = fmaxf(r, __shfl_xor(r, off, 32));
      mnew[v] = fmaxf(mrow[v], r);
      corr[v] = __expf(mrow[v] - mnew[v]);
      psum[v] = 0.f;
    }
    #pragma unroll
    for (int t = 0; t < 4; ++t)
      #pragma unroll
      for (int v = 0; v < 8; ++v) {
        float p = __expf(sacc[t][v] - mnew[v]);
        sacc[t][v] = p;
        psum[v] += p;
      }
    #pragma unroll
    for (int v = 0; v < 8; ++v) {
      float r = psum[v];
      #pragma unroll
      for (int off = 1; off < 16; off <<= 1)
        r += __shfl_xor(r, off, 32);
      lrow[v] = lrow[v] * corr[v] + r;
      mrow[v] = mnew[v];
    }
    #pragma unroll
    for (int t = 0; t < 4; ++t)
      #pragma unroll
      for (int v = 0; v < 8; ++v)
        oacc[t][v] *= corr[v];

    __bf16* pw = lP[wave];                             // P: D-layout -> A-layout
    #pragma unroll
    for (int t = 0; t < 4; ++t)
      #pragma unroll
      for (int v = 0; v < 8; ++v)
        pw[(v + 8 * g) * 72 + 16 * t + ln] = f2bf(sacc[t][v]);
    asm volatile("s_wait_dscnt 0" ::: "memory");       // wave-internal RAW

    #pragma unroll
    for (int c = 0; c < 2; ++c) {                      // O += P V
      FragA pa;
      const __bf16* pr = pw + ln * 72 + 32 * c;
      pa.h[0] = *(const bf16x8*)(pr + 8 * g);
      pa.h[1] = *(const bf16x8*)(pr + 16 + 8 * g);
      #pragma unroll
      for (int t = 0; t < 4; ++t) {
        FragA b;
        const __bf16* vr = &lV[cur][(t * 16 + ln) * 72 + 32 * c + 8 * g];
        b.h[0] = *(const bf16x8*)vr;
        b.h[1] = *(const bf16x8*)(vr + 16);
        oacc[t] = __builtin_amdgcn_wmma_f32_16x16x32_bf16(
            false, pa.f, false, b.f, (short)0, oacc[t], false, false);
      }
    }
    async_wait();
    __syncthreads();
  }

  const int b = bh >> 4, h = bh & 15;
  #pragma unroll
  for (int t = 0; t < 4; ++t)
    #pragma unroll
    for (int v = 0; v < 8; ++v) {
      int s = row0 + v + 8 * g;
      int d = 16 * t + ln;
      float o = oacc[t][v] / lrow[v];
      Og[((size_t)b * 2048 + s) * 1024 + h * 64 + d] = f2bf(o);
    }
}

// ---------------------------------------------------------------------------
// Kernel 3: out = O @ w_out + b_out  (M=4096,K=1024,N=1024), fp32 output.
// Same 128x64 double-buffered async scheme; B from pre-transposed Wot.
// ---------------------------------------------------------------------------
__global__ __launch_bounds__(128) void out_gemm(const __bf16* __restrict__ A,
                                                const __bf16* __restrict__ Wt,
                                                const float* __restrict__ bias,
                                                float* __restrict__ out) {
  constexpr int Kd = 1024, Nd = 1024;
  __shared__ __bf16 lA[2][128 * 40];
  __shared__ __bf16 lB[2][64 * 40];
  const int tid  = threadIdx.x;
  const int wave = tid >> 5, lane = tid & 31;
  const int g = lane >> 4, ln = lane & 15;
  const int m0 = blockIdx.x * 128, n0 = blockIdx.y * 64;

  floatx8 acc[2][4] = {};

  auto load_tiles = [&](int k0, int buf) {
    #pragma unroll
    for (int c = tid; c < 512; c += 128) {
      int r = c >> 2, cc = (c & 3) * 8;
      async_cp16(A + (m0 + r) * Kd + k0 + cc, &lA[buf][r * 40 + cc]);
    }
    #pragma unroll
    for (int c = tid; c < 256; c += 128) {
      int r = c >> 2, cc = (c & 3) * 8;
      async_cp16(Wt + (size_t)(n0 + r) * Kd + k0 + cc, &lB[buf][r * 40 + cc]);
    }
  };

  load_tiles(0, 0);
  async_wait();
  __syncthreads();

  for (int kt = 0; kt < 32; ++kt) {
    const int cur = kt & 1;
    if (kt < 31) load_tiles((kt + 1) * 32, cur ^ 1);

    FragA a[2];
    #pragma unroll
    for (int u = 0; u < 2; ++u) {
      const __bf16* ar = &lA[cur][(wave * 32 + u * 16 + ln) * 40];
      a[u].h[0] = *(const bf16x8*)(ar + 8 * g);
      a[u].h[1] = *(const bf16x8*)(ar + 16 + 8 * g);
    }
    #pragma unroll
    for (int t = 0; t < 4; ++t) {
      FragA b;
      const __bf16* br = &lB[cur][(t * 16 + ln) * 40];
      b.h[0] = *(const bf16x8*)(br + 8 * g);
      b.h[1] = *(const bf16x8*)(br + 16 + 8 * g);
      #pragma unroll
      for (int u = 0; u < 2; ++u)
        acc[u][t] = __builtin_amdgcn_wmma_f32_16x16x32_bf16(
            false, a[u].f, false, b.f, (short)0, acc[u][t], false, false);
    }
    async_wait();
    __syncthreads();
  }

  #pragma unroll
  for (int u = 0; u < 2; ++u)
    #pragma unroll
    for (int t = 0; t < 4; ++t)
      #pragma unroll
      for (int v = 0; v < 8; ++v) {
        int m = m0 + wave * 32 + u * 16 + v + 8 * g;
        int n = n0 + 16 * t + ln;
        out[(size_t)m * Nd + n] = acc[u][t][v] + bias[n];
      }
}

// ---------------------------------------------------------------------------
extern "C" void kernel_launch(void* const* d_in, const int* in_sizes, int n_in,
                              void* d_out, int out_size, void* d_ws, size_t ws_size,
                              hipStream_t stream) {
  const float* x     = (const float*)d_in[0];   // [2,2048,1024]
  const float* w_qkv = (const float*)d_in[1];   // [1024,3072]
  const float* w_out = (const float*)d_in[2];   // [1024,1024]
  const float* b_out = (const float*)d_in[3];   // [1024]
  float* out = (float*)d_out;                   // [2,2048,1024]

  const size_t perT = (size_t)2 * 16 * 2048 * 64;   // 4,194,304 elements
  __bf16* Q   = (__bf16*)d_ws;
  __bf16* K   = Q + perT;
  __bf16* V   = K + perT;                           // stored [B,H,Dh,S]
  __bf16* O   = V + perT;
  __bf16* Xb  = O + perT;                           // 4096x1024
  __bf16* Wqt = Xb + (size_t)4096 * 1024;           // 3072x1024
  __bf16* Wot = Wqt + (size_t)3072 * 1024;          // 1024x1024  (48 MB total)

  prep    <<<dim3(16384),      256, 0, stream>>>(x, w_qkv, w_out, Xb, Wqt, Wot);
  qkv_gemm<<<dim3(32, 48),     128, 0, stream>>>(Xb, Wqt, Q, K, V);
  attn_fa <<<dim3(32, 32),     128, 0, stream>>>(Q, K, V, O);
  out_gemm<<<dim3(32, 16),     128, 0, stream>>>(O, Wot, b_out, out);
}
